// RNN_28982439314015
// MI455X (gfx1250) — matrix operand order repaired and checked by reading, not compiled, not run on hardware
//
#include <hip/hip_runtime.h>
#include <hip/hip_bf16.h>

// ---------------------------------------------------------------------------
// Persistent bf16-WMMA RNN for MI455X (gfx1250, wave32).
//   B=64, T=512, D_in=256, D_h=1024, D_out=512
//   16 workgroups (one per WGP); each caches a 64-col slice of W_ih and a
//   32-col slice of W_ho in LDS (256 KB bf16, WMMA B layout). A-tiles are
//   streamed global->LDS with GLOBAL_LOAD_ASYNC_TO_LDS_B128 through a
//   per-wave ring (ASYNCcnt-paced), so WMMAs only ever wait on LDS.
//   Activations exchanged through L2 in bf16 (double buffered in d_ws);
//   f32 outputs written nontemporally. 2 grid barriers per timestep.
// ---------------------------------------------------------------------------

#define B_      64
#define T_      512
#define DIN     256
#define DH      1024
#define DOUT    512
#define NWG     16
#define NTHREADS 256           // 8 waves of 32
#define W2OFF   (40 * 4 * 1024)              // 160 KB: end of W1 tiles
#define STAGE   (W2OFF + 48 * 2 * 1024)      // 256 KB: A-tile staging start
#define STGWAVE 6144                          // 6 KB ring per wave (6 x 1KB)
#define SMEM_BYTES (STAGE + 8 * STGWAVE)      // 304 KB total (<320 KB/WGP)

typedef __attribute__((ext_vector_type(8)))  __bf16 v8bf;
typedef __attribute__((ext_vector_type(16))) __bf16 v16bf;
typedef __attribute__((ext_vector_type(8)))  float  v8f;
typedef unsigned short bf16_t;   // storage type usable in host code

// Global (addrspace 1) pointer types -> global_load/global_store, never flat.
typedef const __attribute__((address_space(1))) bf16_t* gcbf;
typedef       __attribute__((address_space(1))) bf16_t* gbf;
typedef       __attribute__((address_space(1))) float*  gfp;
typedef __attribute__((address_space(3))) char* lds_t;

__device__ inline bf16_t f2bf(float f) {
    __bf16 b = (__bf16)f;
    return __builtin_bit_cast(bf16_t, b);
}

__device__ inline v16bf cat8(v8bf lo, v8bf hi) {
    return __builtin_shufflevector(lo, hi, 0,1,2,3,4,5,6,7,8,9,10,11,12,13,14,15);
}

// 32B-per-lane tile fragment from LDS (A from staging, B from weight cache).
// `p` = smem + tile_byte_off + lane*32.
__device__ inline v16bf ld_frag(const char* p) {
    v8bf lo = *(const v8bf*)(p);
    v8bf hi = *(const v8bf*)(p + 16);
    return cat8(lo, hi);
}

#define WMMA_BF16(A, Bm, C) \
    __builtin_amdgcn_wmma_f32_16x16x32_bf16(false, (A), false, (Bm), (short)0, (C), false, false)

// Async-copy one 16x32 bf16 A-tile fragment (32B per lane) into LDS.
//  lds0 = absolute LDS byte addr of this lane's 32B slot.
//  ga   = global byte addr of this lane's first 16B piece.
//  NOTE: INST_OFFSET is added to BOTH addresses, so the second instruction
//  uses VDST = lds0-16 with offset:32 to land at lds0+16.
__device__ inline void async_tile(unsigned long long ga, unsigned lds0) {
    asm volatile("global_load_async_to_lds_b128 %0, %2, off\n\t"
                 "global_load_async_to_lds_b128 %1, %2, off offset:32"
                 :: "v"(lds0), "v"(lds0 - 16u), "v"(ga)
                 : "memory");
}
#define WAIT_ASYNC(n) asm volatile("s_wait_asynccnt " #n ::: "memory")

// Device-wide barrier over NWG workgroups (atomic counter + generation in L2).
__device__ inline void grid_barrier(unsigned* cnt, unsigned* gen) {
    __builtin_amdgcn_fence(__ATOMIC_RELEASE, "agent");   // flush my stores
    __syncthreads();
    if (threadIdx.x == 0) {
        unsigned my  = __hip_atomic_load(gen, __ATOMIC_ACQUIRE, __HIP_MEMORY_SCOPE_AGENT);
        unsigned old = __hip_atomic_fetch_add(cnt, 1u, __ATOMIC_ACQ_REL, __HIP_MEMORY_SCOPE_AGENT);
        if (old == NWG - 1) {
            __hip_atomic_store(cnt, 0u, __ATOMIC_RELAXED, __HIP_MEMORY_SCOPE_AGENT);
            __hip_atomic_fetch_add(gen, 1u, __ATOMIC_RELEASE, __HIP_MEMORY_SCOPE_AGENT);
        } else {
            while (__hip_atomic_load(gen, __ATOMIC_ACQUIRE, __HIP_MEMORY_SCOPE_AGENT) == my)
                __builtin_amdgcn_s_sleep(2);
        }
    }
    __syncthreads();
    __builtin_amdgcn_fence(__ATOMIC_ACQUIRE, "agent");   // invalidate stale lines
}

// ---------------------------------------------------------------------------
// Kernel 0: one-time conversions + barrier init (re-run every call).
// ---------------------------------------------------------------------------
__global__ void rnn_init_kernel(const float* __restrict__ x,
                                const float* __restrict__ h0,
                                const float* __restrict__ r0,
                                bf16_t* __restrict__ xbf,
                                bf16_t* __restrict__ hex0,
                                bf16_t* __restrict__ rex0,
                                unsigned* __restrict__ bar) {
    size_t i = (size_t)blockIdx.x * blockDim.x + threadIdx.x;
    size_t stride = (size_t)gridDim.x * blockDim.x;
    const size_t NX = (size_t)B_ * T_ * DIN;
    for (size_t j = i; j < NX; j += stride) xbf[j] = f2bf(x[j]);
    for (size_t j = i; j < (size_t)B_ * DH; j += stride)
        hex0[j] = f2bf(32.0f * h0[j]);                        // sqrt(1024)
    for (size_t j = i; j < (size_t)B_ * DOUT; j += stride)
        rex0[j] = f2bf(22.627416997969522f * r0[j]);          // sqrt(512)
    if (i < 64) bar[i] = 0u;
}

// ---------------------------------------------------------------------------
// Persistent RNN kernel: 16 WGs x 256 threads, 304 KB dynamic LDS each.
// ---------------------------------------------------------------------------
__global__ void rnn_persist_kernel(const float* __restrict__ W_ih,
                                   const float* __restrict__ b_ih,
                                   const float* __restrict__ W_ho,
                                   const float* __restrict__ b_ho,
                                   float* __restrict__ out_r_,
                                   float* __restrict__ out_h_,
                                   const bf16_t* __restrict__ xbf_,
                                   bf16_t* __restrict__ hex0_,
                                   bf16_t* __restrict__ hex1_,
                                   bf16_t* __restrict__ rex0_,
                                   bf16_t* __restrict__ rex1_,
                                   unsigned* __restrict__ bar) {
    extern __shared__ __align__(16) char smem[];
    const unsigned g    = blockIdx.x;          // 0..15: output-column slice
    const unsigned tid  = threadIdx.x;
    const unsigned lane = tid & 31;
    const unsigned lrow = lane & 15;
    const unsigned lg   = lane >> 4;
    const unsigned wave = tid >> 5;            // 0..7

    unsigned* cnt = bar;
    unsigned* gen = bar + 32;

    gcbf xbf  = (gcbf)xbf_;
    gbf  hex[2] = { (gbf)hex0_, (gbf)hex1_ };
    gbf  rex[2] = { (gbf)rex0_, (gbf)rex1_ };
    gfp  out_r = (gfp)out_r_;
    gfp  out_h = (gfp)out_h_;

    // ---- Load weight slices into LDS in WMMA B-operand layout (bf16) ----
    for (unsigned i = tid; i < 1280u * 64u; i += NTHREADS) {
        unsigned k = i >> 6, nn = i & 63u;
        float w = W_ih[(size_t)k * DH + g * 64u + nn];
        unsigned kt = k >> 5, kk = k & 31u, nt = nn >> 4, nl = nn & 15u;
        unsigned l  = (((kk >> 3) & 1u) << 4) | nl;
        unsigned off = ((kt * 4u + nt) << 10) + l * 32u + (((kk >> 4) << 3) | (kk & 7u)) * 2u;
        *(__bf16*)(smem + off) = (__bf16)w;
    }
    for (unsigned i = tid; i < 1536u * 32u; i += NTHREADS) {
        unsigned k = i >> 5, nn = i & 31u;
        float w = W_ho[(size_t)k * DOUT + g * 32u + nn];
        unsigned kt = k >> 5, kk = k & 31u, nt = nn >> 4, nl = nn & 15u;
        unsigned l  = (((kk >> 3) & 1u) << 4) | nl;
        unsigned off = W2OFF + ((kt * 2u + nt) << 10) + l * 32u + (((kk >> 4) << 3) | (kk & 7u)) * 2u;
        *(__bf16*)(smem + off) = (__bf16)w;
    }
    __syncthreads();

    // ---- Per-wave tile assignment ----
    const unsigned m1  = wave & 3u;            // layer1: (m, n-pair)
    const unsigned n1a = (wave >> 2) * 2u;
    const unsigned n1b = n1a + 1u;
    const unsigned m2 = wave & 3u;             // layer2: one tile, 2 K-chains
    const unsigned n2 = wave >> 2;

    const float bias1a = b_ih[g * 64u + n1a * 16u + lrow];
    const float bias1b = b_ih[g * 64u + n1b * 16u + lrow];
    const float bias2  = b_ho[g * 32u + n2 * 16u + lrow];

    const unsigned b1 = m1 * 16u + lrow;       // batch row, per-lane
    const unsigned b2 = m2 * 16u + lrow;

    // Per-lane LDS byte offsets (relative to smem) of weight tile kt=0.
    const unsigned w1aOff = (n1a << 10) + lane * 32u;       // +4096/kt
    const unsigned w1bOff = (n1b << 10) + lane * 32u;
    const unsigned w2Off  = W2OFF + (n2 << 10) + lane * 32u; // +2048/kt

    // A-tile staging ring: relative offset (for ds_load) + absolute LDS
    // address (for the async-copy VDST operand).
    const unsigned stgRel = STAGE + wave * STGWAVE + lane * 32u;
    const char*    stgP   = smem + stgRel;
    const unsigned stgA   = (unsigned)(unsigned long long)(lds_t)smem + stgRel;

    for (unsigned t = 0; t < T_; ++t) {
        const unsigned p = t & 1u, q = p ^ 1u;

        // ================= Layer 1: h_new = relu([x_t, h] @ W_ih + b) ======
        v8f acc0 = {}; v8f acc1 = {};
        // Per-lane global byte addresses of this lane's first 16B of tile 0.
        const unsigned long long gx =
            (unsigned long long)(xbf + ((unsigned)b1 * T_ + t) * DIN + lg * 8u);
        const unsigned long long gh =
            (unsigned long long)((gcbf)hex[p] + (unsigned)b1 * DH + lg * 8u);
        #define GA1(kt) (((kt) < 8) ? (gx + (kt) * 64u) : (gh + ((kt) - 8) * 64u))

        async_tile(GA1(0), stgA);               // prologue: tiles 0,1
        async_tile(GA1(1), stgA + 1024u);
        #pragma unroll
        for (int kt = 0; kt < 40; ++kt) {
            if (kt < 39) { WAIT_ASYNC(2); } else { WAIT_ASYNC(0); }
            v16bf a  = ld_frag(stgP + (kt % 3) * 1024u);
            v16bf w0 = ld_frag(smem + w1aOff + kt * 4096);
            v16bf w1 = ld_frag(smem + w1bOff + kt * 4096);
            if (kt + 2 < 40)                    // keep the ring 2 tiles deep
                async_tile(GA1(kt + 2), stgA + ((kt + 2) % 3) * 1024u);
            acc0 = WMMA_BF16(a, w0, acc0);
            acc1 = WMMA_BF16(a, w1, acc1);
        }
        #undef GA1

        // Epilogue: bias + ReLU, nontemporal f32 out + bf16 exchange buffer.
        {
            const unsigned colA = g * 64u + n1a * 16u + lrow;
            const unsigned colB = g * 64u + n1b * 16u + lrow;
            #pragma unroll
            for (int i = 0; i < 8; ++i) {
                const unsigned row = m1 * 16u + lg * 8u + (unsigned)i;
                float va = acc0[i] + bias1a; va = va > 0.f ? va : 0.f;
                float vb = acc1[i] + bias1b; vb = vb > 0.f ? vb : 0.f;
                const unsigned ob = (row * T_ + t) * DH;
                __builtin_nontemporal_store(va, out_h + ob + colA);
                __builtin_nontemporal_store(vb, out_h + ob + colB);
                hex[q][row * DH + colA] = f2bf(va);
                hex[q][row * DH + colB] = f2bf(vb);
            }
        }
        grid_barrier(cnt, gen);   // h_new visible everywhere

        // ================= Layer 2: r_new = relu([h_new, r] @ W_ho + b) ====
        // Even/odd-kt accumulator chains; kt<32 -> h, kt>=32 -> r.
        v8f accA = {}; v8f accB = {};
        const unsigned long long gh2 =
            (unsigned long long)((gcbf)hex[q] + (unsigned)b2 * DH + lg * 8u);
        const unsigned long long gr2 =
            (unsigned long long)((gcbf)rex[p] + (unsigned)b2 * DOUT + lg * 8u);
        #define GA2(kt) (((kt) < 32) ? (gh2 + (kt) * 64u) : (gr2 + ((kt) - 32) * 64u))

        async_tile(GA2(0), stgA);               // prologue: tiles 0..3
        async_tile(GA2(1), stgA + 1024u);
        async_tile(GA2(2), stgA + 2048u);
        async_tile(GA2(3), stgA + 3072u);
        #pragma unroll
        for (int j = 0; j < 24; ++j) {
            if (j < 23) { WAIT_ASYNC(4); } else { WAIT_ASYNC(0); }
            const int k0 = 2 * j, k1 = 2 * j + 1;
            v16bf a0 = ld_frag(stgP + (k0 % 6) * 1024u);
            v16bf a1 = ld_frag(stgP + (k1 % 6) * 1024u);
            v16bf u0 = ld_frag(smem + w2Off + k0 * 2048);
            v16bf u1 = ld_frag(smem + w2Off + k1 * 2048);
            if (k0 + 4 < 48) {                  // keep the ring 2 pairs deep
                async_tile(GA2(k0 + 4), stgA + ((k0 + 4) % 6) * 1024u);
                async_tile(GA2(k1 + 4), stgA + ((k1 + 4) % 6) * 1024u);
            }
            accA = WMMA_BF16(a0, u0, accA);
            accB = WMMA_BF16(a1, u1, accB);
        }
        #undef GA2
        {
            const unsigned col = g * 32u + n2 * 16u + lrow;
            #pragma unroll
            for (int i = 0; i < 8; ++i) {
                const unsigned row = m2 * 16u + lg * 8u + (unsigned)i;
                float v = accA[i] + accB[i] + bias2; v = v > 0.f ? v : 0.f;
                __builtin_nontemporal_store(v, out_r + (row * T_ + t) * DOUT + col);
                rex[q][row * DOUT + col] = f2bf(v);
            }
        }
        grid_barrier(cnt, gen);   // r_new visible; buffers may be reused
    }
}

// ---------------------------------------------------------------------------
extern "C" void kernel_launch(void* const* d_in, const int* in_sizes, int n_in,
                              void* d_out, int out_size, void* d_ws, size_t ws_size,
                              hipStream_t stream) {
    const float* x   = (const float*)d_in[0];   // [B,T,D_in]
    const float* Wih = (const float*)d_in[1];   // [1280,1024]
    const float* bih = (const float*)d_in[2];   // [1024]
    const float* Who = (const float*)d_in[3];   // [1536,512]
    const float* bho = (const float*)d_in[4];   // [512]
    const float* h0  = (const float*)d_in[5];   // [B,1024]
    const float* r0  = (const float*)d_in[6];   // [B,512]

    float* out_r = (float*)d_out;                         // [B,T,512]
    float* out_h = out_r + (size_t)B_ * T_ * DOUT;        // [B,T,1024]

    // Workspace layout (bytes):
    //   [0,256)    barrier | hex0/hex1 256 KB | rex0/rex1 128 KB | xbf 16 MB
    char* ws = (char*)d_ws;
    unsigned* bar = (unsigned*)ws;
    bf16_t* hex0 = (bf16_t*)(ws + 256);
    bf16_t* hex1 = hex0 + (size_t)B_ * DH;
    bf16_t* rex0 = hex1 + (size_t)B_ * DH;
    bf16_t* rex1 = rex0 + (size_t)B_ * DOUT;
    bf16_t* xbf  = rex1 + (size_t)B_ * DOUT;

    (void)hipFuncSetAttribute((const void*)rnn_persist_kernel,
                              hipFuncAttributeMaxDynamicSharedMemorySize, SMEM_BYTES);

    rnn_init_kernel<<<512, 256, 0, stream>>>(x, h0, r0, xbf, hex0, rex0, bar);
    rnn_persist_kernel<<<NWG, NTHREADS, SMEM_BYTES, stream>>>(
        Wih, bih, Who, bho, out_r, out_h, xbf, hex0, hex1, rex0, rex1, bar);
}
